// loraSpatialSelfAttention_7249904795775
// MI455X (gfx1250) — matrix operand order repaired and checked by reading, not compiled, not run on hardware
//
#include <hip/hip_runtime.h>
#include <hip/hip_bf16.h>
#include <math.h>

// ---------------------------------------------------------------------------
// LoRA spatial self-attention for MI455X (gfx1250, wave32, WMMA + TDM).
//   B=4, C=512, H=W=64 (HW=4096), GROUPS=32, LoRA rank R=4.
// f16 WMMA GEMMs w/ fp32 accum, LoRA folded into effective weights,
// flash-style online-softmax attention (no 256MB S buffer). The f16 working
// set (~80MB) is L2-resident (192MB), so GEMM fragments load straight from
// global; attention V tiles are staged into LDS by the Tensor Data Mover
// (async, overlapped with the QK^T WMMAs, fenced with s_wait_tensorcnt).
// ---------------------------------------------------------------------------

#define BATCH 4
#define CDIM  512
#define HWDIM 4096
#define GROUPS 32
#define CPG   (CDIM / GROUPS)          // 16
#define RANK  4
#define BN    64                        // attention key-block

typedef __attribute__((ext_vector_type(16))) _Float16 v16h;
typedef __attribute__((ext_vector_type(8)))  float    v8f;
typedef __attribute__((ext_vector_type(4)))  unsigned int v4u;
typedef __attribute__((ext_vector_type(8)))  int      v8i;
typedef __attribute__((ext_vector_type(4)))  int      v4i;

__device__ __forceinline__ v8f wmma_f16(v16h a, v16h b, v8f c) {
  // D = A(16x32 f16) * B(32x16 f16) + C(16x16 f32)
  return __builtin_amdgcn_wmma_f32_16x16x32_f16(false, a, false, b,
                                                (short)0, c, false, false);
}

// ---- fragment loaders (wave32 16-bit WMMA, contiguous k order k = 16*(lane>=16) + e in A and B) ----
// A 16x32 (MxK), source row-major [m][k], row stride ld (halfs):
//   lane: m = lane&15; elements 0..15 -> k = k0..k0+15,
//   k0 = (lane>=16) ? 16 : 0.   (two contiguous 16B reads per lane)
__device__ __forceinline__ v16h load_a16x32(const _Float16* base, int ld, int lane) {
  const int m  = lane & 15;
  const int k0 = (lane & 16) ? 16 : 0;
  const _Float16* p = base + m * ld + k0;
  v16h a;
  *reinterpret_cast<uint4*>(&a)       = *reinterpret_cast<const uint4*>(p);
  *(reinterpret_cast<uint4*>(&a) + 1) = *reinterpret_cast<const uint4*>(p + 8);
  return a;
}
// B 32x16 (KxN) where the source is stored N-major: row n holds K contiguous
// (i.e. source is [n][k], row stride ld). lane: n = lane&15;
// elements 0..15 -> k = k0..k0+15, k0 = (lane>=16) ? 16 : 0.
__device__ __forceinline__ v16h load_bT16x32(const _Float16* base, int ld, int lane) {
  const int n  = lane & 15;
  const int k0 = (lane & 16) ? 16 : 0;
  const _Float16* p = base + n * ld + k0;
  v16h b;
  *reinterpret_cast<uint4*>(&b)       = *reinterpret_cast<const uint4*>(p);
  *(reinterpret_cast<uint4*>(&b) + 1) = *reinterpret_cast<const uint4*>(p + 8);
  return b;
}
// B 32x16 (KxN), source stored K-major [k][n] with row stride ld (halfs).
__device__ __forceinline__ v16h load_bKN16x32(const _Float16* base, int ld, int lane) {
  const int n  = lane & 15;
  const int k0 = (lane & 16) ? 16 : 0;
  v16h b;
#pragma unroll
  for (int e = 0; e < 16; ++e) b[e] = base[(k0 + e) * ld + n];
  return b;
}
// C/D 16x16 f32: reg r holds (m = r + 8*(lane>=16), n = lane&15).

// ---------------------------------------------------------------------------
// Tensor Data Mover: async 2D tile load Global -> LDS (ISA ch. 7/8, TDM D#).
// Descriptor: data_size=2B, tensor 512 x 4096 halfs (row stride 512),
// tile = BN rows x 512 halfs starting at gsrc, dest = ldsdst (contiguous).
// This toolchain exposes the 6-arg builtin form:
//   (u32x4 group0, i32x8 group1, i32x4 group2, i32x4 group3, i32x8, i32 cpol)
// ---------------------------------------------------------------------------
#if __has_builtin(__builtin_amdgcn_tensor_load_to_lds)
#define HAVE_TDM 1
__device__ __forceinline__ void tdm_load_v_tile(const _Float16* gsrc, void* ldsdst) {
  const unsigned long long ga = (unsigned long long)(uintptr_t)gsrc;
  const unsigned int la = (unsigned int)(uintptr_t)ldsdst;   // LDS byte offset
  v4u g0;
  g0[0] = 1u;                                   // count=1 (valid), no gather
  g0[1] = la;                                   // lds_addr
  g0[2] = (unsigned int)ga;                     // global_addr[31:0]
  g0[3] = ((unsigned int)(ga >> 32) & 0x01FFFFFFu) | 0x80000000u; // addr[56:32] | type=2
  v8i g1;
  g1[0] = 0x00010000;   // workgroup_mask=0, data_size=1 (2B), no pad/iterate
  g1[1] = 0x02000000;   // tensor_dim0[15:0]=512 in [31:16]
  g1[2] = 0x10000000;   // tensor_dim0[31:16]=0 | tensor_dim1[15:0]=4096 in [31:16]
  g1[3] = 0x02000000;   // tensor_dim1[31:16]=0 | tile_dim0=512 in [31:16]
  g1[4] = BN;           // tile_dim1=BN rows, tile_dim2=0
  g1[5] = CDIM;         // tensor_dim0_stride[31:0]=512
  g1[6] = 0;            // stride[47:32]=0, tensor_dim1_stride[15:0]=0
  g1[7] = 0;
  v4i gz4; gz4[0] = 0; gz4[1] = 0; gz4[2] = 0; gz4[3] = 0;
  v8i gz8; gz8[0] = 0; gz8[1] = 0; gz8[2] = 0; gz8[3] = 0;
  gz8[4] = 0; gz8[5] = 0; gz8[6] = 0; gz8[7] = 0;
  __builtin_amdgcn_tensor_load_to_lds(g0, g1, gz4, gz4, gz8, 0);
}
#else
#define HAVE_TDM 0
#endif

// ---------------------------------------------------------------------------
// Kernel 1: fold LoRA into effective f16 weights.
//   w16all[0]=Wq+Bq*Aq, [1]=Wk+Bk*Ak, [2]=Wv, [3]=Wp  (each 512x512, [o][c])
// ---------------------------------------------------------------------------
__global__ void prep_weights(const float* __restrict__ Wq, const float* __restrict__ Wk,
                             const float* __restrict__ Wv, const float* __restrict__ Wp,
                             const float* __restrict__ Aq, const float* __restrict__ Bq,
                             const float* __restrict__ Ak, const float* __restrict__ Bk,
                             _Float16* __restrict__ w16all) {
  const int mat = blockIdx.y;
  const int idx = blockIdx.x * 256 + threadIdx.x;   // o*512 + c
  const int o = idx >> 9, c = idx & 511;
  float v;
  if (mat == 0) {
    v = Wq[idx];
#pragma unroll
    for (int r = 0; r < RANK; ++r) v += Bq[o * RANK + r] * Aq[r * CDIM + c];
  } else if (mat == 1) {
    v = Wk[idx];
#pragma unroll
    for (int r = 0; r < RANK; ++r) v += Bk[o * RANK + r] * Ak[r * CDIM + c];
  } else if (mat == 2) {
    v = Wv[idx];
  } else {
    v = Wp[idx];
  }
  w16all[(size_t)mat * CDIM * CDIM + idx] = (_Float16)v;
}

// Kernel 2: effective biases.  biasbuf[0]=bq+Bq_b+Bq*Aq_b, [1]=..k.., [2]=bv, [3]=bp
__global__ void prep_bias(const float* __restrict__ bq, const float* __restrict__ bk,
                          const float* __restrict__ bv, const float* __restrict__ bp,
                          const float* __restrict__ Aq_b, const float* __restrict__ Bq_w,
                          const float* __restrict__ Bq_b, const float* __restrict__ Ak_b,
                          const float* __restrict__ Bk_w, const float* __restrict__ Bk_b,
                          float* __restrict__ biasbuf) {
  for (int ch = threadIdx.x; ch < CDIM; ch += blockDim.x) {
    float q = bq[ch] + Bq_b[ch];
    float k = bk[ch] + Bk_b[ch];
#pragma unroll
    for (int r = 0; r < RANK; ++r) {
      q += Bq_w[ch * RANK + r] * Aq_b[r];
      k += Bk_w[ch * RANK + r] * Ak_b[r];
    }
    biasbuf[0 * CDIM + ch] = q;
    biasbuf[1 * CDIM + ch] = k;
    biasbuf[2 * CDIM + ch] = bv[ch];
    biasbuf[3 * CDIM + ch] = bp[ch];
  }
}

// ---------------------------------------------------------------------------
// Kernel 3: GroupNorm over x[b][c][hw]; write token-major f16 h[b][hw][c]
// (token-major so GEMM A-fragments are contiguous over the K=channel dim).
// One block per (batch, group): reduces 16 ch * 4096 px = 65536 elements.
// ---------------------------------------------------------------------------
__global__ void groupnorm_kernel(const float* __restrict__ x,
                                 const float* __restrict__ gamma,
                                 const float* __restrict__ beta,
                                 _Float16* __restrict__ h16) {
  const int b = blockIdx.x >> 5;
  const int g = blockIdx.x & 31;
  const int tid = threadIdx.x;
  const float* xp = x + ((size_t)b * CDIM + g * CPG) * HWDIM;

  float s = 0.f, ss = 0.f;
  for (int idx = tid; idx < CPG * HWDIM; idx += 256) {
    float v = xp[idx];
    s += v;
    ss += v * v;
  }
  __shared__ float r1[256], r2[256];
  r1[tid] = s; r2[tid] = ss;
  __syncthreads();
  for (int off = 128; off > 0; off >>= 1) {
    if (tid < off) { r1[tid] += r1[tid + off]; r2[tid] += r2[tid + off]; }
    __syncthreads();
  }
  const float inv_n = 1.f / (float)(CPG * HWDIM);
  const float mean = r1[0] * inv_n;
  const float var  = r2[0] * inv_n - mean * mean;
  const float rstd = rsqrtf(var + 1e-6f);

  for (int idx = tid; idx < CPG * HWDIM; idx += 256) {
    const int cl = idx >> 12;        // channel-in-group
    const int hw = idx & (HWDIM - 1);
    const int ch = g * CPG + cl;
    float v = (xp[idx] - mean) * rstd * gamma[ch] + beta[ch];
    h16[((size_t)b * HWDIM + hw) * CDIM + ch] = (_Float16)v;
  }
}

// ---------------------------------------------------------------------------
// Kernel 4: Q/K/V projection GEMM.  Y[b][hw][co] = h[b][hw][:] @ Weff[co][:]^T + bias
// grid: x = HW/64 (M), y = C/32 (N), z = b*3 + which. block = 256 (8 waves,
// wave grid 4M x 2N, 16x16 WMMA tile each, K=512 chained in 16 steps).
// Weights + h are L2-resident, so fragments load straight from global.
// ---------------------------------------------------------------------------
__global__ void qkv_gemm(const _Float16* __restrict__ h16,
                         const _Float16* __restrict__ w16all,
                         const float* __restrict__ biasbuf,
                         _Float16* __restrict__ qbuf,
                         _Float16* __restrict__ kbuf,
                         _Float16* __restrict__ vbuf) {
  const int lane  = threadIdx.x & 31;
  const int wid   = threadIdx.x >> 5;
  const int b     = blockIdx.z / 3;
  const int which = blockIdx.z % 3;
  const int row0  = blockIdx.x * 64 + (wid & 3) * 16;
  const int col0  = blockIdx.y * 32 + (wid >> 2) * 16;

  const _Float16* A  = h16 + ((size_t)b * HWDIM + row0) * CDIM;
  const _Float16* Bw = w16all + (size_t)which * CDIM * CDIM + (size_t)col0 * CDIM;
  _Float16* out = (which == 0) ? qbuf : (which == 1) ? kbuf : vbuf;

  v8f acc = {};
#pragma unroll
  for (int kk = 0; kk < CDIM; kk += 32) {
    __builtin_prefetch(A + kk + 128, 0, 3);        // global_prefetch_b8 (near, RT)
    v16h a  = load_a16x32(A + kk, CDIM, lane);
    v16h bf = load_bT16x32(Bw + kk, CDIM, lane);
    acc = wmma_f16(a, bf, acc);
  }

  const int n = col0 + (lane & 15);
  const float bias = biasbuf[which * CDIM + n];
#pragma unroll
  for (int r = 0; r < 8; ++r) {
    const int m = row0 + r + ((lane & 16) ? 8 : 0);
    out[((size_t)b * HWDIM + m) * CDIM + n] = (_Float16)(acc[r] + bias);
  }
}

// ---------------------------------------------------------------------------
// Kernel 5: flash attention.  grid: x = HW/16 query-blocks, y = batch.
// block = 128 (4 waves). Wave w owns head-dim chunk c0=[128w,128w+128):
//   - split-K partial S = Q Kᵀ over its chunk (WMMA), reduced via LDS
//   - online softmax (rows owned by tid<16), P tile in LDS (f16)
//   - O_chunk += P @ V_chunk (WMMA), V tile staged in LDS by the TDM,
//     overlapped with the S WMMAs and fenced with s_wait_tensorcnt.
// ---------------------------------------------------------------------------
__global__ void attn_flash(const _Float16* __restrict__ Q,
                           const _Float16* __restrict__ K,
                           const _Float16* __restrict__ V,
                           _Float16* __restrict__ O) {
  __shared__ alignas(16) _Float16 vtile[BN][CDIM];     // 64 KB
  __shared__ float  sred[4][16][BN];                   // 16 KB split-K partials
  __shared__ alignas(16) _Float16 pbuf[16][BN];        // 2 KB  exp(S-m) tile
  __shared__ float  mrow[16], lrow[16], corr[16];

  const int tid  = threadIdx.x;
  const int lane = tid & 31;
  const int w    = tid >> 5;          // wave id -> head-dim chunk
  const int b    = blockIdx.y;
  const int i0   = blockIdx.x * 16;   // query block
  const int c0   = w * 128;
  const float scale = rsqrtf((float)CDIM);

  // Q fragments for this wave's head-dim chunk (resident in VGPRs all along)
  v16h aq[4];
#pragma unroll
  for (int kt = 0; kt < 4; ++kt)
    aq[kt] = load_a16x32(Q + ((size_t)b * HWDIM + i0) * CDIM + c0 + kt * 32, CDIM, lane);

  v8f oacc[8] = {};                   // 16 x 128 fp32 output accumulator
  if (tid < 16) { mrow[tid] = -3.0e38f; lrow[tid] = 0.f; }

  for (int t = 0; t < HWDIM / BN; ++t) {
    const int j0 = t * BN;

    // (A) all waves done consuming vtile/pbuf/sred of previous iteration
    __syncthreads();

    // kick the async V-tile DMA (wave 0 only; TENSORcnt tracks completion)
    const _Float16* vsrc = V + ((size_t)b * HWDIM + j0) * CDIM;
#if HAVE_TDM
    if (w == 0) tdm_load_v_tile(vsrc, &vtile[0][0]);
#endif

    // --- partial S = Q[:, c0:c0+128] * K[:, c0:c0+128]^T (global K reads,
    //     overlaps with the TDM transfer) ---
    v8f sp[4] = {};
#pragma unroll
    for (int nt = 0; nt < 4; ++nt) {
      const _Float16* kb = K + ((size_t)b * HWDIM + j0 + nt * 16) * CDIM + c0;
#pragma unroll
      for (int kt = 0; kt < 4; ++kt)
        sp[nt] = wmma_f16(aq[kt], load_bT16x32(kb + kt * 32, CDIM, lane), sp[nt]);
    }

    // scatter partial S to LDS for the cross-wave split-K reduction
    const int ihalf = (lane & 16) ? 8 : 0;
#pragma unroll
    for (int nt = 0; nt < 4; ++nt)
#pragma unroll
      for (int r = 0; r < 8; ++r)
        sred[w][r + ihalf][nt * 16 + (lane & 15)] = sp[nt][r];

#if HAVE_TDM
    if (w == 0) __builtin_amdgcn_s_wait_tensorcnt(0);   // V tile landed in LDS
#else
    for (int idx = tid; idx < BN * CDIM / 8; idx += 128) {
      const int row = idx >> 6;            // CDIM/8 = 64 uint4 per row
      const int col = (idx & 63) * 8;
      *reinterpret_cast<uint4*>(&vtile[row][col]) =
          *reinterpret_cast<const uint4*>(vsrc + (size_t)row * CDIM + col);
    }
#endif
    __syncthreads();   // (B) sred + vtile visible to everyone

    // --- online softmax over this 16xBN block (rows owned by tid<16) ---
    if (tid < 16) {
      const int l = tid;
      float m_old = mrow[l];
      float mb = -3.0e38f;
      float srow[BN];
#pragma unroll
      for (int j = 0; j < BN; ++j) {
        float sv = scale * (sred[0][l][j] + sred[1][l][j] + sred[2][l][j] + sred[3][l][j]);
        srow[j] = sv;
        mb = fmaxf(mb, sv);
      }
      const float mnew = fmaxf(m_old, mb);
      const float cf = __expf(m_old - mnew);
      float psum = 0.f;
#pragma unroll
      for (int j = 0; j < BN; ++j) {
        float p = __expf(srow[j] - mnew);
        psum += p;
        pbuf[l][j] = (_Float16)p;
      }
      mrow[l] = mnew;
      lrow[l] = lrow[l] * cf + psum;
      corr[l] = cf;
    }
    __syncthreads();   // (C) pbuf/corr ready

    // --- rescale accumulator, then O += P @ V_chunk ---
#pragma unroll
    for (int r = 0; r < 8; ++r) {
      const float cf = corr[r + ihalf];
#pragma unroll
      for (int nt = 0; nt < 8; ++nt) oacc[nt][r] *= cf;
    }
#pragma unroll
    for (int kt = 0; kt < 2; ++kt) {
      v16h ap = load_a16x32(&pbuf[0][kt * 32], BN, lane);
#pragma unroll
      for (int nt = 0; nt < 8; ++nt)
        oacc[nt] = wmma_f16(ap, load_bKN16x32(&vtile[kt * 32][c0 + nt * 16], CDIM, lane),
                            oacc[nt]);
    }
  }

  // --- normalize by row sums, store O[b][hw][c] f16 ---
  const int ihalf = (lane & 16) ? 8 : 0;
#pragma unroll
  for (int r = 0; r < 8; ++r) {
    const float linv = 1.f / lrow[r + ihalf];
    const int m = i0 + r + ihalf;
#pragma unroll
    for (int nt = 0; nt < 8; ++nt)
      O[((size_t)b * HWDIM + m) * CDIM + c0 + nt * 16 + (lane & 15)] =
          (_Float16)(oacc[nt][r] * linv);
  }
}

// ---------------------------------------------------------------------------
// Kernel 6: output projection + residual.  out[b][c][hw] = x + O @ Wp^T + bp
// ---------------------------------------------------------------------------
__global__ void proj_residual(const _Float16* __restrict__ O16,
                              const _Float16* __restrict__ w16all,
                              const float* __restrict__ biasbuf,
                              const float* __restrict__ x,
                              float* __restrict__ out) {
  const int lane = threadIdx.x & 31;
  const int wid  = threadIdx.x >> 5;
  const int b    = blockIdx.z;
  const int row0 = blockIdx.x * 64 + (wid & 3) * 16;
  const int col0 = blockIdx.y * 32 + (wid >> 2) * 16;

  const _Float16* A  = O16 + ((size_t)b * HWDIM + row0) * CDIM;
  const _Float16* Bw = w16all + (size_t)3 * CDIM * CDIM + (size_t)col0 * CDIM;

  v8f acc = {};
#pragma unroll
  for (int kk = 0; kk < CDIM; kk += 32) {
    __builtin_prefetch(A + kk + 128, 0, 3);
    v16h a  = load_a16x32(A + kk, CDIM, lane);
    v16h bf = load_bT16x32(Bw + kk, CDIM, lane);
    acc = wmma_f16(a, bf, acc);
  }

  const int n = col0 + (lane & 15);
  const float bias = biasbuf[3 * CDIM + n];
#pragma unroll
  for (int r = 0; r < 8; ++r) {
    const int m = row0 + r + ((lane & 16) ? 8 : 0);
    const size_t oi = ((size_t)b * CDIM + n) * HWDIM + m;
    out[oi] = acc[r] + bias + x[oi];
  }
}

// ---------------------------------------------------------------------------
// Workspace layout (bytes). Total ~86 MB (fp16 tensors are L2-resident).
// ---------------------------------------------------------------------------
#define WS_W16    ((size_t)0)                              // 4*512*512*2 = 2 MB
#define WS_BIAS   (WS_W16  + (size_t)4 * CDIM * CDIM * 2)  // 4*512*4     = 8 KB
#define WS_H16    (WS_BIAS + (size_t)4 * CDIM * 4)         // 16 MB each below
#define TOK_BYTES ((size_t)BATCH * HWDIM * CDIM * 2)
#define WS_Q      (WS_H16 + TOK_BYTES)
#define WS_K      (WS_Q   + TOK_BYTES)
#define WS_V      (WS_K   + TOK_BYTES)
#define WS_O      (WS_V   + TOK_BYTES)

extern "C" void kernel_launch(void* const* d_in, const int* in_sizes, int n_in,
                              void* d_out, int out_size, void* d_ws, size_t ws_size,
                              hipStream_t stream) {
  const float* x    = (const float*)d_in[0];
  const float* gn_w = (const float*)d_in[1];
  const float* gn_b = (const float*)d_in[2];
  const float* Wq   = (const float*)d_in[3];
  const float* bq   = (const float*)d_in[4];
  const float* Wk   = (const float*)d_in[5];
  const float* bk   = (const float*)d_in[6];
  const float* Wv   = (const float*)d_in[7];
  const float* bv   = (const float*)d_in[8];
  const float* Wp   = (const float*)d_in[9];
  const float* bp   = (const float*)d_in[10];
  const float* Aq_w = (const float*)d_in[11];
  const float* Aq_b = (const float*)d_in[12];
  const float* Bq_w = (const float*)d_in[13];
  const float* Bq_b = (const float*)d_in[14];
  const float* Ak_w = (const float*)d_in[15];
  const float* Ak_b = (const float*)d_in[16];
  const float* Bk_w = (const float*)d_in[17];
  const float* Bk_b = (const float*)d_in[18];
  float* out = (float*)d_out;

  char* ws = (char*)d_ws;
  _Float16* w16all  = (_Float16*)(ws + WS_W16);
  float*    biasbuf = (float*)   (ws + WS_BIAS);
  _Float16* h16     = (_Float16*)(ws + WS_H16);
  _Float16* qbuf    = (_Float16*)(ws + WS_Q);
  _Float16* kbuf    = (_Float16*)(ws + WS_K);
  _Float16* vbuf    = (_Float16*)(ws + WS_V);
  _Float16* obuf    = (_Float16*)(ws + WS_O);

  prep_weights<<<dim3(CDIM * CDIM / 256, 4), 256, 0, stream>>>(
      Wq, Wk, Wv, Wp, Aq_w, Bq_w, Ak_w, Bk_w, w16all);
  prep_bias<<<1, 256, 0, stream>>>(
      bq, bk, bv, bp, Aq_b, Bq_w, Bq_b, Ak_b, Bk_w, Bk_b, biasbuf);
  groupnorm_kernel<<<BATCH * GROUPS, 256, 0, stream>>>(x, gn_w, gn_b, h16);
  qkv_gemm<<<dim3(HWDIM / 64, CDIM / 32, BATCH * 3), 256, 0, stream>>>(
      h16, w16all, biasbuf, qbuf, kbuf, vbuf);
  attn_flash<<<dim3(HWDIM / 16, BATCH), 128, 0, stream>>>(qbuf, kbuf, vbuf, obuf);
  proj_residual<<<dim3(HWDIM / 64, CDIM / 32, BATCH), 256, 0, stream>>>(
      obuf, w16all, biasbuf, x, out);
}
